// QuadConvLayer_39616778338559
// MI455X (gfx1250) — matrix operand order repaired and checked by reading, not compiled, not run on hardware
//
#include <hip/hip_runtime.h>

typedef __attribute__((ext_vector_type(16))) _Float16 v16h;
typedef __attribute__((ext_vector_type(8)))  float    v8f;

#define QC_Bn     4
#define QC_CIN    8
#define QC_COUT   8
#define QC_PIN    16384
#define QC_POUT   4096
#define QC_NEVAL  1000000
#define QC_HID    64
#define QC_DECAY  1048576.0f   // (P_IN/16)^2

#define WAVES   4      // waves per block (128 threads)
#define ITERS   8      // point-groups per wave per block
#define APITCH  72     // halves per row of h tile (16B-aligned rows, padded)
#define FPITCH  68     // floats per row of filt tile (272B rows, 16B-aligned)
#define GPITCH  36     // floats per row of gathered tile (144B rows, 16B-aligned)

union Frag16 { v16h v; uint4 q[2]; };

__global__ void __launch_bounds__(128) qc_zero(float* __restrict__ out, int n) {
  int i = blockIdx.x * blockDim.x + threadIdx.x;
  if (i < n) out[i] = 0.0f;
}

__global__ void __launch_bounds__(128) qc_main(
    const float* __restrict__ features,   // (4,8,16384)
    const float* __restrict__ locs,       // (N,2)
    const float* __restrict__ qw,         // (16384,)
    const float* __restrict__ W0,         // (2,64)
    const float* __restrict__ W1,         // (64,64)
    const float* __restrict__ W2,         // (64,64)
    const int*   __restrict__ idx_out,    // (N,) sorted
    const int*   __restrict__ idx_in,     // (N,)
    float*       __restrict__ out)        // (4,8,4096)
{
  // B-fragment-swizzled f16 weights (staging only; resident in VGPRs after load)
  __shared__ __align__(16) _Float16 w1s[8 * 32 * 16];
  __shared__ __align__(16) _Float16 w2s[8 * 32 * 16];
  __shared__ float w0s[2 * QC_HID];
  __shared__ __align__(16) _Float16 atile[WAVES][16 * APITCH]; // h tile (f16)
  __shared__ __align__(16) float    ftile[WAVES][16 * FPITCH]; // filt^T tile (f32, [p][j*8+i])
  __shared__ __align__(16) float    gtile[WAVES][16 * GPITCH]; // gathered features
  __shared__ float wvw [WAVES][16];   // quadrature weight * bump
  __shared__ int   wvii[WAVES][16];   // idx_in
  __shared__ int   wvio[WAVES][16];   // idx_out
  __shared__ float wvx [WAVES][16];
  __shared__ float wvy [WAVES][16];

  const int tid  = threadIdx.x;
  const int wv   = tid >> 5;
  const int lane = tid & 31;
  const int lr   = lane & 15;   // row / col-in-tile
  const int lg   = lane >> 4;   // lane group (K striping)

  // ---- one-time: swizzle W1/W2 into B-fragment layout (f16), stage W0 ----
  for (int f = wv; f < 8; f += WAVES) {
    const int t = f >> 1, kb = f & 1;
    const int col = t * 16 + lr;
    #pragma unroll
    for (int e = 0; e < 16; ++e) {
      const int K = kb * 32 + (e < 8 ? e + 8 * lg : 8 + e + 8 * lg);   // B-matrix: e<8 -> K = e+8g ; e>=8 -> K = 8+e+8g
      w1s[(f * 32 + lane) * 16 + e] = (_Float16)W1[K * QC_HID + col];
      w2s[(f * 32 + lane) * 16 + e] = (_Float16)W2[K * QC_HID + col];
    }
  }
  if (tid < 2 * QC_HID) w0s[tid] = W0[tid];
  __syncthreads();

  // ---- hoist ALL loop-invariant B fragments into registers (16 x v16h) ----
  Frag16 bw1[8], bw2[8];
  #pragma unroll
  for (int f = 0; f < 8; ++f) {
    const _Float16* p1 = &w1s[(f * 32 + lane) * 16];
    const _Float16* p2 = &w2s[(f * 32 + lane) * 16];
    bw1[f].q[0] = *reinterpret_cast<const uint4*>(p1);
    bw1[f].q[1] = *reinterpret_cast<const uint4*>(p1 + 8);
    bw2[f].q[0] = *reinterpret_cast<const uint4*>(p2);
    bw2[f].q[1] = *reinterpret_cast<const uint4*>(p2 + 8);
  }

  for (int it = 0; it < ITERS; ++it) {
    const long long gidx = (long long)blockIdx.x * (WAVES * ITERS) + it * WAVES + wv;
    const long long n0 = gidx * 16;

    // ---- per-point scalars: locs, bump weight, indices (+ next-iter prefetch) ----
    if (lane < 16) {
      const long long n = n0 + lane;
      float x = 0.f, y = 0.f, w = 0.f;
      int ii = 0, io = 0;
      if (n < QC_NEVAL) {
        x = locs[2 * n]; y = locs[2 * n + 1];
        ii = idx_in[n];  io = idx_out[n];
        const float r2 = x * x + y * y;
        const float d  = 1.0f - QC_DECAY * (r2 * r2);
        const float bump = (d > 1e-12f) ? __expf(1.0f - 1.0f / d) : 0.0f;
        w = bump * qw[ii];
      }
      // prefetch next iteration's streams (this wave advances by WAVES*16 pts)
      const long long np = n + (long long)WAVES * 16;
      if (it + 1 < ITERS && np < QC_NEVAL) {
        __builtin_prefetch(&locs[2 * np], 0, 1);
        __builtin_prefetch(&idx_in[np],  0, 1);
        __builtin_prefetch(&idx_out[np], 0, 1);
      }
      wvx[wv][lane] = x;  wvy[wv][lane] = y;
      wvw[wv][lane] = w;  wvii[wv][lane] = ii;  wvio[wv][lane] = io;
    }
    __syncthreads();

    // ---- stage 0: h0 = sin(locs @ W0), f16 A-tile (each lane: 32 cols of row lr) ----
    {
      const float x = wvx[wv][lr], y = wvy[wv][lr];
      _Float16* arow = &atile[wv][lr * APITCH + lg * 32];
      #pragma unroll
      for (int k = 0; k < 32; k += 2) {
        const int c = lg * 32 + k;
        const float v0 = __sinf(x * w0s[c]     + y * w0s[QC_HID + c]);
        const float v1 = __sinf(x * w0s[c + 1] + y * w0s[QC_HID + c + 1]);
        union { _Float16 h[2]; unsigned u; } pk;
        pk.h[0] = (_Float16)v0; pk.h[1] = (_Float16)v1;
        *reinterpret_cast<unsigned*>(arow + k) = pk.u;
      }
    }
    // ---- gather features for the 16 points (L2-resident table) -> b128 stores ----
    {
      const int ii = wvii[wv][lr];
      const float* fbase = features + ii;
      float* grow = &gtile[wv][lr * GPITCH + lg * 16];
      #pragma unroll
      for (int blk = 0; blk < 4; ++blk) {
        float4 v;
        v.x = fbase[(lg * 16 + blk * 4 + 0) * QC_PIN];
        v.y = fbase[(lg * 16 + blk * 4 + 1) * QC_PIN];
        v.z = fbase[(lg * 16 + blk * 4 + 2) * QC_PIN];
        v.w = fbase[(lg * 16 + blk * 4 + 3) * QC_PIN];
        *reinterpret_cast<float4*>(grow + blk * 4) = v;
      }
    }
    __syncthreads();

    // ---- GEMM1: h1 = sin(h0 @ W1) ----
    // Phase A: 8 back-to-back WMMAs into 4 independent accumulators.
    // Phase B: sin + f16-convert + column stores as a separate pass.
    {
      Frag16 a0, a1;
      const _Float16* ar = &atile[wv][lr * APITCH];
      // A-matrix: e<8 -> K = e+8g ; e>=8 -> K = 8+e+8g  => two b128 loads per frag
      a0.q[0] = *reinterpret_cast<const uint4*>(ar + 8 * lg);
      a0.q[1] = *reinterpret_cast<const uint4*>(ar + 16 + 8 * lg);
      a1.q[0] = *reinterpret_cast<const uint4*>(ar + 32 + 8 * lg);
      a1.q[1] = *reinterpret_cast<const uint4*>(ar + 48 + 8 * lg);
      v8f c[4];
      #pragma unroll
      for (int t = 0; t < 4; ++t) {
        v8f z = {0.f, 0.f, 0.f, 0.f, 0.f, 0.f, 0.f, 0.f};
        c[t] = __builtin_amdgcn_wmma_f32_16x16x32_f16(false, a0.v, false, bw1[t*2+0].v, (short)0, z,    false, false);
      }
      #pragma unroll
      for (int t = 0; t < 4; ++t)
        c[t] = __builtin_amdgcn_wmma_f32_16x16x32_f16(false, a1.v, false, bw1[t*2+1].v, (short)0, c[t], false, false);
      #pragma unroll
      for (int t = 0; t < 4; ++t) {
        #pragma unroll
        for (int r = 0; r < 8; ++r)     // D: row M = r + 8g, col N = lr
          atile[wv][(r + 8 * lg) * APITCH + t * 16 + lr] = (_Float16)__sinf(c[t][r]);
      }
    }
    __syncthreads();

    // ---- GEMM2: filt = h1 @ W2  (WMMA -> transposed f32 tile [p][j*8+i]) ----
    {
      Frag16 a0, a1;
      const _Float16* ar = &atile[wv][lr * APITCH];
      a0.q[0] = *reinterpret_cast<const uint4*>(ar + 8 * lg);
      a0.q[1] = *reinterpret_cast<const uint4*>(ar + 16 + 8 * lg);
      a1.q[0] = *reinterpret_cast<const uint4*>(ar + 32 + 8 * lg);
      a1.q[1] = *reinterpret_cast<const uint4*>(ar + 48 + 8 * lg);
      v8f c[4];
      #pragma unroll
      for (int t = 0; t < 4; ++t) {
        v8f z = {0.f, 0.f, 0.f, 0.f, 0.f, 0.f, 0.f, 0.f};
        c[t] = __builtin_amdgcn_wmma_f32_16x16x32_f16(false, a0.v, false, bw2[t*2+0].v, (short)0, z,    false, false);
      }
      #pragma unroll
      for (int t = 0; t < 4; ++t)
        c[t] = __builtin_amdgcn_wmma_f32_16x16x32_f16(false, a1.v, false, bw2[t*2+1].v, (short)0, c[t], false, false);
      #pragma unroll
      for (int t = 0; t < 4; ++t) {
        const int cc = t * 16 + lr;     // original col = i*8 + j
        const int tr = (cc & 7) * 8 + (cc >> 3);   // store transposed: j*8 + i
        #pragma unroll
        for (int r = 0; r < 8; ++r)
          ftile[wv][(r + 8 * lg) * FPITCH + tr] = c[t][r];
      }
    }
    __syncthreads();

    // ---- einsum + sorted segment-sum (vector LDS reads, run-compressed atomics) ----
    {
      const int b = lane >> 3;           // 0..3
      const int j = lane & 7;            // 0..7
      float* outbj = out + (b * QC_COUT + j) * QC_POUT;
      int cur = -1; float run = 0.f;
      #pragma unroll
      for (int p = 0; p < 16; ++p) {
        const float* fr = &ftile[wv][p * FPITCH + j * 8];      // filt^T: i contiguous
        const float* gr = &gtile[wv][p * GPITCH + b * QC_CIN]; // features: i contiguous
        const float4 g0 = *reinterpret_cast<const float4*>(gr);
        const float4 g1 = *reinterpret_cast<const float4*>(gr + 4);
        const float4 f0 = *reinterpret_cast<const float4*>(fr);
        const float4 f1 = *reinterpret_cast<const float4*>(fr + 4);
        float acc = g0.x * f0.x + g0.y * f0.y + g0.z * f0.z + g0.w * f0.w
                  + g1.x * f1.x + g1.y * f1.y + g1.z * f1.z + g1.w * f1.w;
        const float val = wvw[wv][p] * acc;
        const int io = wvio[wv][p];
        if (io == cur) { run += val; }
        else {
          if (cur >= 0) atomicAdd(outbj + cur, run);
          cur = io; run = val;
        }
      }
      if (cur >= 0) atomicAdd(outbj + cur, run);
    }
    __syncthreads();
  }
}

extern "C" void kernel_launch(void* const* d_in, const int* in_sizes, int n_in,
                              void* d_out, int out_size, void* d_ws, size_t ws_size,
                              hipStream_t stream) {
  (void)in_sizes; (void)n_in; (void)d_ws; (void)ws_size;
  const float* features = (const float*)d_in[0];
  const float* locs     = (const float*)d_in[1];
  const float* qwv      = (const float*)d_in[2];
  const float* W0       = (const float*)d_in[3];
  const float* W1       = (const float*)d_in[4];
  const float* W2       = (const float*)d_in[5];
  const int*   idx_out  = (const int*)d_in[6];
  const int*   idx_in   = (const int*)d_in[7];
  float* out = (float*)d_out;

  // atomics accumulate into d_out: zero it every call (harness poisons once)
  qc_zero<<<(out_size + 127) / 128, 128, 0, stream>>>(out, out_size);

  const int groups    = (QC_NEVAL + 15) / 16;     // 62500 point-groups of 16
  const int per_block = WAVES * ITERS;            // 32 groups per block
  const int blocks    = (groups + per_block - 1) / per_block;   // 1954
  qc_main<<<blocks, 128, 0, stream>>>(features, locs, qwv, W0, W1, W2,
                                      idx_out, idx_in, out);
}
